// LanguageModel_76338748719434
// MI455X (gfx1250) — matrix operand order repaired and checked by reading, not compiled, not run on hardware
//
#include <hip/hip_runtime.h>

// ---------------- problem constants ----------------
#define BB 16
#define TT 1024
#define EE 512
#define HH 8
#define LL 6
#define VV 32
#define HDIM 64
#define MM (BB * TT) // 16384 token rows

typedef __bf16 bf16;
typedef __bf16 v16bf __attribute__((ext_vector_type(16)));
typedef float  v8f   __attribute__((ext_vector_type(8)));
typedef unsigned int u32x4 __attribute__((ext_vector_type(4)));
typedef float  f32x4 __attribute__((ext_vector_type(4)));

union Frag { v16bf bf; u32x4 q[2]; };

static __device__ __forceinline__ v8f wmma_bf16(const Frag &a, const Frag &b, v8f c) {
  return __builtin_amdgcn_wmma_f32_16x16x32_bf16(false, a.bf, false, b.bf,
                                                 (short)0, c, false, false);
}

// Async global -> LDS copy of 16 bytes per lane (CDNA5, tracked by ASYNCcnt).
// ldsoff: LDS byte address (VGPR). base: uniform 64-bit base (SGPR pair).
// voff: per-lane unsigned byte offset (VGPR). GVS mode: mem = saddr + vaddr.
static __device__ __forceinline__ void async_copy_b128(unsigned ldsoff,
                                                       const void *base,
                                                       unsigned voff) {
  asm volatile("global_load_async_to_lds_b128 %0, %1, %2"
               :: "v"(ldsoff), "v"(voff), "s"(base) : "memory");
}

// ---------------- weight packing (fp32 -> bf16, TRANSPOSED to [N,K]) ----------------
// QKV packed as WT[l][(sel*512 + h*64 + d) * E + e] so B-tiles are contiguous rows.
__global__ __launch_bounds__(256) void pack_qkv_kernel(
    const float *__restrict__ Wq, const float *__restrict__ Wk,
    const float *__restrict__ Wv, bf16 *__restrict__ dst) {
  int idx = blockIdx.x * 256 + threadIdx.x;
  if (idx >= LL * HH * EE * HDIM) return;
  int d = idx & 63;
  int e = (idx >> 6) & 511;
  int h = (idx >> 15) & 7;
  int l = idx >> 18;
  size_t src = ((size_t)(l * HH + h) * EE + e) * HDIM + d;
  size_t lbase = (size_t)l * (3 * EE) * EE;
  size_t n = (size_t)h * HDIM + d;
  dst[lbase + (n) * EE + e]            = (bf16)Wq[src];
  dst[lbase + (n + EE) * EE + e]       = (bf16)Wk[src];
  dst[lbase + (n + 2 * EE) * EE + e]   = (bf16)Wv[src];
}

// src: Lc stacked [K,N] fp32 row-major; dst: Lc stacked [N,K] bf16 row-major.
__global__ __launch_bounds__(256) void transpose_bf16_kernel(
    const float *__restrict__ src, bf16 *__restrict__ dst, int K, int N, int Lc) {
  int idx = blockIdx.x * 256 + threadIdx.x;
  if (idx >= Lc * K * N) return;
  int l = idx / (K * N);
  int rem = idx - l * K * N;
  int k = rem / N, n = rem - k * N;
  dst[(size_t)l * K * N + (size_t)n * K + k] = (bf16)src[idx];
}

// ---------------- embedding ----------------
__global__ __launch_bounds__(128) void embed_kernel(
    const int *__restrict__ tok, const float *__restrict__ emb,
    const float *__restrict__ pos, float *__restrict__ x) {
  int row = blockIdx.x;            // b*T + t
  int t = row & (TT - 1);
  int tk = tok[row];
  int c = threadIdx.x * 4;
  f32x4 e = *(const f32x4 *)&emb[(size_t)tk * EE + c];
  f32x4 p = *(const f32x4 *)&pos[(size_t)t * EE + c];
  *(f32x4 *)&x[(size_t)row * EE + c] = e + p;
}

// ---------------- layernorm (fp32 in -> bf16 out), one wave per row ----------------
__global__ __launch_bounds__(256) void ln_kernel(
    const float *__restrict__ x, const float *__restrict__ g,
    const float *__restrict__ b, bf16 *__restrict__ out) {
  int wave = threadIdx.x >> 5, lane = threadIdx.x & 31;
  int row = blockIdx.x * 8 + wave;
  const f32x4 *xr = (const f32x4 *)(x + (size_t)row * EE + lane * 16);
  union { f32x4 v[4]; float e[16]; } u;
#pragma unroll
  for (int i = 0; i < 4; i++) u.v[i] = xr[i];
  float s = 0.f, q = 0.f;
#pragma unroll
  for (int i = 0; i < 16; i++) { s += u.e[i]; q += u.e[i] * u.e[i]; }
#pragma unroll
  for (int off = 16; off >= 1; off >>= 1) {
    s += __shfl_xor(s, off, 32);
    q += __shfl_xor(q, off, 32);
  }
  float mu = s * (1.f / EE);
  float var = q * (1.f / EE) - mu * mu;
  float rs = rsqrtf(var + 1e-5f);
  int c0 = lane * 16;
#pragma unroll
  for (int i = 0; i < 16; i++) {
    int c = c0 + i;
    out[(size_t)row * EE + c] = (bf16)((u.e[i] - mu) * rs * g[c] + b[c]);
  }
}

// ---------------- tiled bf16 WMMA GEMM, async double-buffered LDS staging ----------
// C[M,N] = act(A[M,K] * WT[N,K]^T + bias + res). A bf16 row-major (lda=K),
// WT bf16 row-major [N,K]. Outputs row-major (ldc=N). M%128==0, K%32==0.
// Column-guarded for N < 64 (logits): OOB lanes dump into an LDS bin row.
template <bool BIAS, bool RELU, bool RES, bool OUTF, bool OUTB>
__global__ __launch_bounds__(256) void gemm_bf16_kernel(
    const bf16 *__restrict__ A, const bf16 *__restrict__ WT,
    const float *__restrict__ bias, const float *__restrict__ res,
    float *__restrict__ outf, bf16 *__restrict__ outb, int M, int N, int K) {
  __shared__ __align__(16) bf16 As[2][128][40];
  __shared__ __align__(16) bf16 Bs[2][65][40]; // row 64 = garbage bin for OOB lanes
  int tid = threadIdx.x;
  int wave = tid >> 5, lane = tid & 31;
  int wm = wave & 3, wn = wave >> 2;      // 4x2 wave grid -> 128x64 tile
  int hl = lane >> 4, l16 = lane & 15;
  int row0 = blockIdx.y * 128, n0 = blockIdx.x * 64;

  unsigned baseA = (unsigned)(uintptr_t)&As[0][0][0];
  unsigned baseB = (unsigned)(uintptr_t)&Bs[0][0][0];
  const unsigned bufStrideA = 128 * 40 * 2;
  const unsigned bufStrideB = 65 * 40 * 2;

  // zero pad region once if this block's N-tile is partial (logits GEMM)
  if (n0 + 64 > N) {
    for (int i = tid; i < 2 * 65 * 40; i += 256) ((bf16 *)Bs)[i] = (bf16)0.f;
  }

  // A tile: 128 rows x 64B -> 512 x 16B chunks, 2 per thread.
  // B tile: 64 rows x 64B -> 256 x 16B chunks, 1 per thread.
  int ar0 = tid >> 1;                 // chunk pair base: rows tid/2 w/ 2 chunks
  int ac0 = (tid & 1) * 2;            // chunk index 0/2 within row (of 4)
  int bn = tid >> 2, bc = tid & 3;    // B row / chunk
  bool bvalid = (n0 + bn) < N;
  unsigned bLds0 = baseB + (bvalid ? (unsigned)(bn * 80 + bc * 16) : (unsigned)(64 * 80));
  unsigned bGOff0 = bvalid ? (unsigned)(((size_t)(n0 + bn) * K) * 2 + bc * 16) : 0u;

  auto issue_tiles = [&](int k0, int buf) {
    unsigned aLds = baseA + buf * bufStrideA + ar0 * 80 + ac0 * 16;
    unsigned aG = (unsigned)(((size_t)(row0 + ar0) * K + k0) * 2 + ac0 * 16);
    async_copy_b128(aLds, A, aG);
    async_copy_b128(aLds + 16, A, aG + 16);
    async_copy_b128(bLds0 + buf * bufStrideB, WT, bGOff0 + (unsigned)(k0 * 2));
  };

  v8f acc[2][2] = {};
  int nsteps = K >> 5;
  issue_tiles(0, 0);
  for (int i = 0; i < nsteps; i++) {
    int buf = i & 1;
    if (i + 1 < nsteps) {
      issue_tiles((i + 1) << 5, buf ^ 1);
      asm volatile("s_wait_asynccnt 0x3" ::: "memory"); // step-i copies done
    } else {
      asm volatile("s_wait_asynccnt 0x0" ::: "memory");
    }
    __syncthreads();
    Frag a[2], b[2];
#pragma unroll
    for (int mi = 0; mi < 2; mi++) { // A layout: k = (i/8)*16 + hl*8 + i%8
      int r = wm * 32 + mi * 16 + l16;
      a[mi].q[0] = *(const u32x4 *)&As[buf][r][hl * 8];
      a[mi].q[1] = *(const u32x4 *)&As[buf][r][16 + hl * 8];
    }
#pragma unroll
    for (int ni = 0; ni < 2; ni++) { // B layout: k = (i/8)*16 + hl*8 + i%8
      int nn = wn * 32 + ni * 16 + l16;
      b[ni].q[0] = *(const u32x4 *)&Bs[buf][nn][hl * 8];
      b[ni].q[1] = *(const u32x4 *)&Bs[buf][nn][16 + hl * 8];
    }
#pragma unroll
    for (int mi = 0; mi < 2; mi++)
#pragma unroll
      for (int ni = 0; ni < 2; ni++)
        acc[mi][ni] = wmma_bf16(a[mi], b[ni], acc[mi][ni]);
    __syncthreads(); // all waves done reading before buffer is overwritten
  }
  // epilogue: D layout -> row = r + 8*hl, col = l16
#pragma unroll
  for (int mi = 0; mi < 2; mi++)
#pragma unroll
    for (int ni = 0; ni < 2; ni++) {
      int gn = n0 + wn * 32 + ni * 16 + l16;
      if (gn < N) {
#pragma unroll
        for (int r = 0; r < 8; r++) {
          int gm = row0 + wm * 32 + mi * 16 + r + 8 * hl;
          float v = acc[mi][ni][r];
          if (BIAS) v += bias[gn];
          if (RES) v += res[(size_t)gm * N + gn];
          if (RELU) v = fmaxf(v, 0.f);
          if (OUTF) outf[(size_t)gm * N + gn] = v;
          if (OUTB) outb[(size_t)gm * N + gn] = (bf16)v;
        }
      }
    }
}

// ---------------- flash attention (causal, no 1/sqrt(d) per reference) ----------------
// qkv: bf16 [M, 1536] (q | k | v, each head-concat h*64+d). out: bf16 [M, 512].
// One wave per 16-row query tile; key pairs of 32 so P*V is one K=32 WMMA.
__global__ __launch_bounds__(256) void attn_kernel(
    const bf16 *__restrict__ qkv, bf16 *__restrict__ out) {
  __shared__ __align__(16) bf16 Plds[8][16][40];
  __shared__ __align__(16) bf16 Vlds[8][64][40]; // transposed: [d][kk]
  int w = threadIdx.x >> 5, lane = threadIdx.x & 31;
  int hl = lane >> 4, l16 = lane & 15;
  int idx = blockIdx.x * 8 + w;  // B*H*(T/16) = 8192 tiles
  int it = idx & 63;             // T/16 = 64
  int bh = idx >> 6;
  int b = bh >> 3, h = bh & 7;
  int qi = it * 16;
  const int LD = 3 * EE; // 1536

  // Q tile as two 16x32 A-fragments (d = f*32 + (i/8)*16 + hl*8 + i%8)
  Frag qa[2];
  {
    const bf16 *qr = qkv + (size_t)(b * TT + qi + l16) * LD + h * HDIM;
#pragma unroll
    for (int f = 0; f < 2; f++) {
      qa[f].q[0] = *(const u32x4 *)&qr[f * 32 + hl * 8];
      qa[f].q[1] = *(const u32x4 *)&qr[f * 32 + 16 + hl * 8];
    }
  }

  float mrow[8], lrow[8];
  v8f o[4] = {};
#pragma unroll
  for (int r = 0; r < 8; r++) { mrow[r] = -1e30f; lrow[r] = 0.f; }

  int jjmax = (qi + 15) >> 5;
  for (int jj = 0; jj <= jjmax; ++jj) {
    int kb = jj * 32;
    // ---- S = Q * K^T for two 16-key tiles ----
    Frag k0f[2], k1f[2];
    {
      const bf16 *kr0 = qkv + (size_t)(b * TT + kb + l16) * LD + EE + h * HDIM;
      const bf16 *kr1 = kr0 + (size_t)16 * LD;
#pragma unroll
      for (int s = 0; s < 2; s++) { // B layout: kk = (i/8)*16 + hl*8 + i%8, d = s*32 + kk
        k0f[s].q[0] = *(const u32x4 *)&kr0[s * 32 + hl * 8];
        k0f[s].q[1] = *(const u32x4 *)&kr0[s * 32 + 16 + hl * 8];
        k1f[s].q[0] = *(const u32x4 *)&kr1[s * 32 + hl * 8];
        k1f[s].q[1] = *(const u32x4 *)&kr1[s * 32 + 16 + hl * 8];
      }
    }
    v8f s0 = {}, s1 = {};
    s0 = wmma_bf16(qa[0], k0f[0], s0);
    s0 = wmma_bf16(qa[1], k0f[1], s0);
    s1 = wmma_bf16(qa[0], k1f[0], s1);
    s1 = wmma_bf16(qa[1], k1f[1], s1);

    // ---- causal mask + online softmax (8 rows per lane, 16-lane row groups) ----
    float p0[8], p1[8], scl[8];
#pragma unroll
    for (int r = 0; r < 8; r++) {
      int qr = qi + r + 8 * hl;
      float v0 = (kb + l16 <= qr) ? s0[r] : -1e30f;
      float v1 = (kb + 16 + l16 <= qr) ? s1[r] : -1e30f;
      float mx = fmaxf(v0, v1);
#pragma unroll
      for (int off = 8; off >= 1; off >>= 1) mx = fmaxf(mx, __shfl_xor(mx, off, 32));
      float mn = fmaxf(mrow[r], mx);
      float sc = __expf(mrow[r] - mn);
      float e0 = __expf(v0 - mn), e1 = __expf(v1 - mn);
      float sm = e0 + e1;
#pragma unroll
      for (int off = 8; off >= 1; off >>= 1) sm += __shfl_xor(sm, off, 32);
      lrow[r] = lrow[r] * sc + sm;
      mrow[r] = mn;
      p0[r] = e0; p1[r] = e1; scl[r] = sc;
    }
#pragma unroll
    for (int ni = 0; ni < 4; ni++)
#pragma unroll
      for (int r = 0; r < 8; r++) o[ni][r] *= scl[r];

    // ---- restage P (D layout -> A layout) and V (row -> transposed) via LDS ----
#pragma unroll
    for (int r = 0; r < 8; r++) {
      int row = r + 8 * hl;
      Plds[w][row][l16]      = (bf16)p0[r];
      Plds[w][row][16 + l16] = (bf16)p1[r];
    }
    {
      const bf16 *vr = qkv + (size_t)(b * TT + kb + lane) * LD + 2 * EE + h * HDIM;
      union { u32x4 q[4]; bf16 e[64]; } vv;
#pragma unroll
      for (int c = 0; c < 4; c++) vv.q[c] = *(const u32x4 *)&vr[c * 16];
#pragma unroll
      for (int d = 0; d < 64; d++) Vlds[w][d][lane] = vv.e[d];
    }
    asm volatile("s_wait_dscnt 0" ::: "memory"); // same-wave LDS WB->RD ordering

    Frag pf;
    pf.q[0] = *(const u32x4 *)&Plds[w][l16][hl * 8];
    pf.q[1] = *(const u32x4 *)&Plds[w][l16][16 + hl * 8];
#pragma unroll
    for (int ni = 0; ni < 4; ni++) {
      Frag vf;
      vf.q[0] = *(const u32x4 *)&Vlds[w][ni * 16 + l16][hl * 8];
      vf.q[1] = *(const u32x4 *)&Vlds[w][ni * 16 + l16][16 + hl * 8];
      o[ni] = wmma_bf16(pf, vf, o[ni]);
    }
  }

  // ---- normalize + store (bf16, head-concat layout) ----
#pragma unroll
  for (int ni = 0; ni < 4; ni++)
#pragma unroll
    for (int r = 0; r < 8; r++) {
      int row = qi + r + 8 * hl;
      out[(size_t)(b * TT + row) * EE + h * HDIM + ni * 16 + l16] =
          (bf16)(o[ni][r] * (1.f / lrow[r]));
    }
}

// ---------------- orchestration ----------------
extern "C" void kernel_launch(void *const *d_in, const int *in_sizes, int n_in,
                              void *d_out, int out_size, void *d_ws, size_t ws_size,
                              hipStream_t stream) {
  (void)in_sizes; (void)n_in; (void)out_size; (void)ws_size;
  const int   *tok   = (const int *)d_in[0];
  const float *emb   = (const float *)d_in[1];
  const float *pos   = (const float *)d_in[2];
  const float *Wq    = (const float *)d_in[3];
  const float *Wk    = (const float *)d_in[4];
  const float *Wv    = (const float *)d_in[5];
  const float *Wo    = (const float *)d_in[6];
  const float *bo    = (const float *)d_in[7];
  const float *ln1g  = (const float *)d_in[8];
  const float *ln1b  = (const float *)d_in[9];
  const float *ln2g  = (const float *)d_in[10];
  const float *ln2b  = (const float *)d_in[11];
  const float *W1    = (const float *)d_in[12];
  const float *b1    = (const float *)d_in[13];
  const float *W2    = (const float *)d_in[14];
  const float *b2    = (const float *)d_in[15];
  const float *lnfg  = (const float *)d_in[16];
  const float *lnfb  = (const float *)d_in[17];
  const float *Wf    = (const float *)d_in[18];
  const float *bfv   = (const float *)d_in[19];
  float *logits = (float *)d_out;

  char *ws = (char *)d_ws;
  size_t off = 0;
  auto alloc = [&](size_t bytes) -> char * {
    char *p = ws + off;
    off += (bytes + 255) & ~(size_t)255;
    return p;
  };
  float *X    = (float *)alloc((size_t)MM * EE * 4);          // residual stream
  bf16 *Hb    = (bf16 *)alloc((size_t)MM * EE * 2);           // LN output
  bf16 *QKV   = (bf16 *)alloc((size_t)MM * 3 * EE * 2);
  bf16 *ATT   = (bf16 *)alloc((size_t)MM * EE * 2);
  bf16 *Fb    = (bf16 *)alloc((size_t)MM * EE * 2);
  bf16 *WQKVt = (bf16 *)alloc((size_t)LL * EE * 3 * EE * 2);  // [3E, E] per layer
  bf16 *WOt   = (bf16 *)alloc((size_t)LL * EE * EE * 2);      // [E, E] transposed
  bf16 *W1t   = (bf16 *)alloc((size_t)LL * EE * EE * 2);
  bf16 *W2t   = (bf16 *)alloc((size_t)LL * EE * EE * 2);
  bf16 *WFt   = (bf16 *)alloc((size_t)EE * VV * 2);           // [V, E]

  // weight packing (transposed bf16)
  pack_qkv_kernel<<<(LL * HH * EE * HDIM + 255) / 256, 256, 0, stream>>>(Wq, Wk, Wv, WQKVt);
  int nEE = LL * EE * EE;
  transpose_bf16_kernel<<<(nEE + 255) / 256, 256, 0, stream>>>(Wo, WOt, EE, EE, LL);
  transpose_bf16_kernel<<<(nEE + 255) / 256, 256, 0, stream>>>(W1, W1t, EE, EE, LL);
  transpose_bf16_kernel<<<(nEE + 255) / 256, 256, 0, stream>>>(W2, W2t, EE, EE, LL);
  transpose_bf16_kernel<<<(EE * VV + 255) / 256, 256, 0, stream>>>(Wf, WFt, EE, VV, 1);

  // x = emb[tokens] + pos
  embed_kernel<<<MM, 128, 0, stream>>>(tok, emb, pos, X);

  dim3 blk(256);
  dim3 gQKV(3 * EE / 64, MM / 128);
  dim3 gE(EE / 64, MM / 128);
  dim3 gV(1, MM / 128);

  for (int l = 0; l < LL; ++l) {
    ln_kernel<<<MM / 8, blk, 0, stream>>>(X, ln1g + l * EE, ln1b + l * EE, Hb);
    gemm_bf16_kernel<false, false, false, false, true><<<gQKV, blk, 0, stream>>>(
        Hb, WQKVt + (size_t)l * EE * 3 * EE, nullptr, nullptr, nullptr, QKV,
        MM, 3 * EE, EE);
    attn_kernel<<<MM / 16 / 8, blk, 0, stream>>>(QKV, ATT);
    gemm_bf16_kernel<true, false, true, true, false><<<gE, blk, 0, stream>>>(
        ATT, WOt + (size_t)l * EE * EE, bo + l * EE, X, X, nullptr, MM, EE, EE);
    ln_kernel<<<MM / 8, blk, 0, stream>>>(X, ln2g + l * EE, ln2b + l * EE, Hb);
    gemm_bf16_kernel<true, true, false, false, true><<<gE, blk, 0, stream>>>(
        Hb, W1t + (size_t)l * EE * EE, b1 + l * EE, nullptr, nullptr, Fb, MM, EE, EE);
    gemm_bf16_kernel<true, false, true, true, false><<<gE, blk, 0, stream>>>(
        Fb, W2t + (size_t)l * EE * EE, b2 + l * EE, X, X, nullptr, MM, EE, EE);
  }

  ln_kernel<<<MM / 8, blk, 0, stream>>>(X, lnfg, lnfb, Hb);
  gemm_bf16_kernel<true, false, false, true, false><<<gV, blk, 0, stream>>>(
      Hb, WFt, bfv, nullptr, logits, nullptr, MM, VV, EE);
}
